// PointwiseConv_90185723281818
// MI455X (gfx1250) — matrix operand order repaired and checked by reading, not compiled, not run on hardware
//
#include <hip/hip_runtime.h>
#include <hip/hip_bf16.h>
#include <stdint.h>

// ---------------------------------------------------------------------------
// Sizes (fixed by the reference)
// ---------------------------------------------------------------------------
#define BATCH 8
#define NPTS  2048
#define CFEAT 128
#define PDIM  3
#define KNN   16
#define CIN   131          // CFEAT + PDIM
#define KP1   160          // CIN padded to multiple of 32 for WMMA K-steps
#define OCH   256
#define MROWS (BATCH * NPTS)   // 16384

typedef _Float16 v16h __attribute__((ext_vector_type(16)));
typedef _Float16 v8h  __attribute__((ext_vector_type(8)));
typedef float    v8f  __attribute__((ext_vector_type(8)));

// ---------------------------------------------------------------------------
// Kernel 1: kNN (top-16 smallest squared distances) per query point.
// One thread per query; per-batch positions cached in LDS (2048*3*4 = 24 KB).
// 16-entry sorted list kept in registers (fully unrolled insertion).
// ---------------------------------------------------------------------------
__global__ __launch_bounds__(256) void knn_kernel(const float* __restrict__ pos,
                                                  int* __restrict__ nn_idx) {
    __shared__ float sx[NPTS], sy[NPTS], sz[NPTS];
    const int b     = blockIdx.x >> 3;   // 8 chunks of 256 queries per batch
    const int chunk = blockIdx.x & 7;
    const float* pb = pos + (size_t)b * NPTS * PDIM;
    for (int i = threadIdx.x; i < NPTS; i += 256) {
        sx[i] = pb[i * 3 + 0];
        sy[i] = pb[i * 3 + 1];
        sz[i] = pb[i * 3 + 2];
    }
    __syncthreads();

    const int q = chunk * 256 + threadIdx.x;
    const float qx = sx[q], qy = sy[q], qz = sz[q];

    float bd[KNN];
    int   bi[KNN];
#pragma unroll
    for (int t = 0; t < KNN; ++t) { bd[t] = 3.4e38f; bi[t] = 0; }

    for (int j = 0; j < NPTS; ++j) {
        const float dx = sx[j] - qx;
        const float dy = sy[j] - qy;
        const float dz = sz[j] - qz;
        const float d  = fmaf(dx, dx, fmaf(dy, dy, dz * dz));
        if (d < bd[KNN - 1]) {              // beats current worst -> insert
            bd[KNN - 1] = d; bi[KNN - 1] = j;
#pragma unroll
            for (int t = KNN - 1; t > 0; --t) {
                if (bd[t] < bd[t - 1]) {    // strict: earlier index wins ties
                    float td = bd[t]; bd[t] = bd[t - 1]; bd[t - 1] = td;
                    int   ti = bi[t]; bi[t] = bi[t - 1]; bi[t - 1] = ti;
                }
            }
        }
    }
    int* o = nn_idx + ((size_t)b * NPTS + q) * KNN;
#pragma unroll
    for (int t = 0; t < KNN; ++t) o[t] = bi[t];
}

// ---------------------------------------------------------------------------
// Kernel 2: gather neighbors of [x || pos], mean over K, write f16 A-matrix
// padded to KP1 columns (zeros in 131..159).  One block per (b, n) row.
// ---------------------------------------------------------------------------
__global__ __launch_bounds__(KP1) void gather_mean_kernel(const float* __restrict__ x,
                                                          const float* __restrict__ pos,
                                                          const int* __restrict__ nn_idx,
                                                          _Float16* __restrict__ avgh) {
    __shared__ int sidx[KNN];
    const int m = blockIdx.x;          // b * NPTS + n
    const int b = m >> 11;             // NPTS = 2048
    if (threadIdx.x < KNN) sidx[threadIdx.x] = nn_idx[(size_t)m * KNN + threadIdx.x];
    __syncthreads();

    const int c = threadIdx.x;
    float s = 0.0f;
    if (c < CFEAT) {
        const float* xb = x + (size_t)b * NPTS * CFEAT;
#pragma unroll
        for (int t = 0; t < KNN; ++t) s += xb[(size_t)sidx[t] * CFEAT + c];
    } else if (c < CIN) {
        const int p = c - CFEAT;
        const float* pb = pos + (size_t)b * NPTS * PDIM;
#pragma unroll
        for (int t = 0; t < KNN; ++t) s += pb[(size_t)sidx[t] * PDIM + p];
    }
    const float v = (c < CIN) ? (s * (1.0f / (float)KNN)) : 0.0f;
    avgh[(size_t)m * KP1 + c] = (_Float16)v;
}

// ---------------------------------------------------------------------------
// f32 [rows_src, cols] -> f16 TRANSPOSED [cols, rows_dst] with zero K padding.
// One-time cost on tiny, L2-resident weight matrices; lets the GEMM stage B
// tiles with contiguous 16B copies and no transpose.
// ---------------------------------------------------------------------------
__global__ void convert_transpose_pad_kernel(const float* __restrict__ src,
                                             _Float16* __restrict__ dst,
                                             int rows_src, int rows_dst, int cols) {
    const int i = blockIdx.x * blockDim.x + threadIdx.x;
    if (i >= cols * rows_dst) return;
    const int n = i / rows_dst;   // output channel
    const int k = i % rows_dst;   // input channel (padded)
    dst[i] = (k < rows_src) ? (_Float16)src[(size_t)k * cols + n] : (_Float16)0.0f;
}

// ---------------------------------------------------------------------------
// WMMA GEMM:  Cout[M,N] = A[M,Kp](f16) * Bt[N,Kp](f16, pre-transposed) + bias.
// Block = 128 threads (4 waves); 64x64 output tile; BK = 32.
// Wave w computes rows [w*16, w*16+16) x all 64 cols via 4 f32 accumulators.
// LDS pitch 40 halves (80 B) -> every fragment chunk is 16B-aligned, so each
// WMMA operand is exactly two ds_load_b128.  Staging is branch-free: 4 batched
// global_load_b128 then 4 ds_store_b128.  Layouts per ISA 7.12.2 (wave32).
// ---------------------------------------------------------------------------
#define LPITCH 40

__global__ __launch_bounds__(128) void gemm_wmma_kernel(const _Float16* __restrict__ A,
                                                        const _Float16* __restrict__ Bt,
                                                        const float* __restrict__ bias,
                                                        float* __restrict__ Cout,
                                                        int M, int Kp, int Nout) {
    __shared__ __align__(16) _Float16 As[64][LPITCH];   // 64 rows x 32 k
    __shared__ __align__(16) _Float16 Bs[64][LPITCH];   // 64 cols x 32 k (k-major)

    const int tid  = threadIdx.x;
    const int wave = tid >> 5;
    const int lane = tid & 31;
    const int m0   = blockIdx.x * 64;
    const int n0   = blockIdx.y * 64;

    // Per-thread staging slot: thread covers rows r and r+32, one 16B k-chunk.
    const int r  = tid >> 2;          // 0..31
    const int kc = (tid & 3) << 3;    // 0,8,16,24

    v8f acc[4] = {};

    for (int k0 = 0; k0 < Kp; k0 += 32) {
        // Stage A (64x32) and Bt (64x32): load 4 chunks, then store 4 chunks.
        const _Float16* gA = &A [(size_t)(m0 + r) * Kp + (k0 + kc)];
        const _Float16* gB = &Bt[(size_t)(n0 + r) * Kp + (k0 + kc)];
        const v8h ra0 = *(const v8h*)gA;
        const v8h ra1 = *(const v8h*)(gA + (size_t)32 * Kp);
        const v8h rb0 = *(const v8h*)gB;
        const v8h rb1 = *(const v8h*)(gB + (size_t)32 * Kp);
        *(v8h*)&As[r][kc]      = ra0;
        *(v8h*)&As[r + 32][kc] = ra1;
        *(v8h*)&Bs[r][kc]      = rb0;
        *(v8h*)&Bs[r + 32][kc] = rb1;
        if (k0 + 32 < Kp) {   // prefetch next K tile (global_prefetch_b8)
            __builtin_prefetch(gA + 32, 0, 0);
            __builtin_prefetch(gB + 32, 0, 0);
        }
        __syncthreads();

        // A fragment: 16x32, row = lane&15; lanes<16 hold K 0..7 & 16..23,
        // lanes>=16 hold K 8..15 & 24..31.  Two ds_load_b128.
        const int arow = (wave << 4) + (lane & 15);
        const int kb   = (lane < 16) ? 0 : 8;
        const v8h a_lo = *(const v8h*)&As[arow][kb];
        const v8h a_hi = *(const v8h*)&As[arow][kb + 16];
        const v16h af  = __builtin_shufflevector(a_lo, a_hi,
                             0, 1, 2, 3, 4, 5, 6, 7, 8, 9, 10, 11, 12, 13, 14, 15);

        // B fragments: 32x16 (K-major column per lane), col = lane&15;
        // lanes<16 hold K 0..7 & 16..23, lanes>=16 hold K 8..15 & 24..31.  Two ds_load_b128.
        const int bcol = lane & 15;
        const int kb2  = (lane < 16) ? 0 : 8;
#pragma unroll
        for (int nt = 0; nt < 4; ++nt) {
            const v8h b_lo = *(const v8h*)&Bs[(nt << 4) + bcol][kb2];
            const v8h b_hi = *(const v8h*)&Bs[(nt << 4) + bcol][kb2 + 16];
            const v16h bf  = __builtin_shufflevector(b_lo, b_hi,
                                 0, 1, 2, 3, 4, 5, 6, 7, 8, 9, 10, 11, 12, 13, 14, 15);
            acc[nt] = __builtin_amdgcn_wmma_f32_16x16x32_f16(
                false, af, false, bf, (short)0, acc[nt], false, false);
        }
        __syncthreads();
    }

    // Epilogue: C/D layout — VGPR r holds M=r (lanes 0-15) / M=r+8 (lanes 16-31)
    const int rbase = (lane < 16) ? 0 : 8;
    const int col0  = lane & 15;
#pragma unroll
    for (int nt = 0; nt < 4; ++nt) {
        const int col = n0 + (nt << 4) + col0;
        const float bv = bias[col];
#pragma unroll
        for (int rr = 0; rr < 8; ++rr) {
            const int row = m0 + (wave << 4) + rbase + rr;
            Cout[(size_t)row * Nout + col] = acc[nt][rr] + bv;
        }
    }
}

// ---------------------------------------------------------------------------
// BN helpers: zero stats, coalesced per-channel sum/sumsq, apply.
// ---------------------------------------------------------------------------
__global__ void zero_kernel(float* __restrict__ p, int n) {
    const int i = blockIdx.x * blockDim.x + threadIdx.x;
    if (i < n) p[i] = 0.0f;
}

__global__ __launch_bounds__(OCH) void colreduce_kernel(const float* __restrict__ src,
                                                        float* __restrict__ gsum,
                                                        float* __restrict__ gsq) {
    const int c  = threadIdx.x;          // channel 0..255, coalesced across lanes
    const int r0 = blockIdx.x * 128;
    float s = 0.0f, q = 0.0f;
    for (int r = 0; r < 128; ++r) {
        const float v = src[(size_t)(r0 + r) * OCH + c];
        s += v;
        q = fmaf(v, v, q);
    }
    atomicAdd(&gsum[c], s);
    atomicAdd(&gsq[c], q);
}

__global__ void bnrelu_f16_kernel(const float* __restrict__ pre,
                                  const float* __restrict__ gsum, const float* __restrict__ gsq,
                                  const float* __restrict__ gamma, const float* __restrict__ beta,
                                  _Float16* __restrict__ out) {
    const int i = blockIdx.x * blockDim.x + threadIdx.x;
    if (i >= MROWS * OCH) return;
    const int c = i & (OCH - 1);
    const float inv = 1.0f / (float)MROWS;
    const float mean = gsum[c] * inv;
    const float var  = gsq[c] * inv - mean * mean;
    const float rs   = rsqrtf(var + 1e-5f);
    float v = (pre[i] - mean) * rs * gamma[c] + beta[c];
    v = fmaxf(v, 0.0f);
    out[i] = (_Float16)v;
}

__global__ void bn_out_kernel(const float* __restrict__ pre,
                              const float* __restrict__ gsum, const float* __restrict__ gsq,
                              const float* __restrict__ gamma, const float* __restrict__ beta,
                              float* __restrict__ out) {
    const int i = blockIdx.x * blockDim.x + threadIdx.x;
    if (i >= MROWS * OCH) return;
    const int c = i & (OCH - 1);
    const float inv = 1.0f / (float)MROWS;
    const float mean = gsum[c] * inv;
    const float var  = gsq[c] * inv - mean * mean;
    const float rs   = rsqrtf(var + 1e-5f);
    out[i] = (pre[i] - mean) * rs * gamma[c] + beta[c];
}

// ---------------------------------------------------------------------------
// Host-side orchestration
// ---------------------------------------------------------------------------
extern "C" void kernel_launch(void* const* d_in, const int* in_sizes, int n_in,
                              void* d_out, int out_size, void* d_ws, size_t ws_size,
                              hipStream_t stream) {
    const float* x   = (const float*)d_in[0];
    const float* pos = (const float*)d_in[1];
    const float* W1  = (const float*)d_in[2];
    const float* b1  = (const float*)d_in[3];
    const float* g1  = (const float*)d_in[4];
    const float* be1 = (const float*)d_in[5];
    const float* W2  = (const float*)d_in[6];
    const float* b2  = (const float*)d_in[7];
    const float* g2  = (const float*)d_in[8];
    const float* be2 = (const float*)d_in[9];
    float* out = (float*)d_out;

    // Carve workspace (all offsets 256B aligned)
    char*  ws  = (char*)d_ws;
    size_t off = 0;
    auto carve = [&](size_t bytes) -> void* {
        void* p = ws + off;
        off += (bytes + 255) & ~(size_t)255;
        return p;
    };
    int*      nn_idx = (int*)     carve((size_t)MROWS * KNN * sizeof(int));       //  1 MB
    _Float16* avgh   = (_Float16*)carve((size_t)MROWS * KP1 * sizeof(_Float16));  //  5 MB
    _Float16* w1t    = (_Float16*)carve((size_t)OCH * KP1 * sizeof(_Float16));    // 80 KB  [OCH, KP1]
    _Float16* w2t    = (_Float16*)carve((size_t)OCH * OCH * sizeof(_Float16));    //128 KB  [OCH, OCH]
    float*    pre    = (float*)   carve((size_t)MROWS * OCH * sizeof(float));     // 16 MB (reused)
    _Float16* hh     = (_Float16*)carve((size_t)MROWS * OCH * sizeof(_Float16));  //  8 MB
    float*    gsum   = (float*)   carve(2 * OCH * sizeof(float));
    float*    gsq    = gsum + OCH;

    const int elems = MROWS * OCH;

    // 1) kNN indices
    knn_kernel<<<dim3(BATCH * (NPTS / 256)), 256, 0, stream>>>(pos, nn_idx);

    // 2) gather + mean -> f16 A-matrix [MROWS, KP1]
    gather_mean_kernel<<<dim3(MROWS), KP1, 0, stream>>>(x, pos, nn_idx, avgh);

    // 3) weights -> f16, transposed to [N, Kp] (W1 K padded 131 -> 160)
    convert_transpose_pad_kernel<<<(OCH * KP1 + 255) / 256, 256, 0, stream>>>(W1, w1t,
                                                                              CIN, KP1, OCH);
    convert_transpose_pad_kernel<<<(OCH * OCH + 255) / 256, 256, 0, stream>>>(W2, w2t,
                                                                              OCH, OCH, OCH);

    // 4) GEMM1 + bias -> pre (f32)
    gemm_wmma_kernel<<<dim3(MROWS / 64, OCH / 64), 128, 0, stream>>>(avgh, w1t, b1, pre,
                                                                     MROWS, KP1, OCH);

    // 5) BN stats + apply + ReLU -> hh (f16)
    zero_kernel<<<2, OCH, 0, stream>>>(gsum, 2 * OCH);
    colreduce_kernel<<<MROWS / 128, OCH, 0, stream>>>(pre, gsum, gsq);
    bnrelu_f16_kernel<<<(elems + 255) / 256, 256, 0, stream>>>(pre, gsum, gsq, g1, be1, hh);

    // 6) GEMM2 + bias -> pre (f32, buffer reused)
    gemm_wmma_kernel<<<dim3(MROWS / 64, OCH / 64), 128, 0, stream>>>(hh, w2t, b2, pre,
                                                                     MROWS, OCH, OCH);

    // 7) BN stats + apply -> out (f32)
    zero_kernel<<<2, OCH, 0, stream>>>(gsum, 2 * OCH);
    colreduce_kernel<<<MROWS / 128, OCH, 0, stream>>>(pre, gsum, gsq);
    bn_out_kernel<<<(elems + 255) / 256, 256, 0, stream>>>(pre, gsum, gsq, g2, be2, out);

    (void)in_sizes; (void)n_in; (void)out_size; (void)ws_size;
}
